// MLPResNetBlock_Pro_33225867002508
// MI455X (gfx1250) — matrix operand order repaired and checked by reading, not compiled, not run on hardware
//
#include <hip/hip_runtime.h>
#include <hip/hip_bf16.h>

// ---------------------------------------------------------------------------
// CDNA5 (gfx1250) WMMA types
// ---------------------------------------------------------------------------
typedef __attribute__((ext_vector_type(16))) __bf16 v16bf;
typedef __attribute__((ext_vector_type(8)))  float  v8f;

union BF16Frag {
    v16bf  v;
    __bf16 e[16];
    uint4  q[2];
};

#define DIM 1024
#define NHEAD 16
#define HDIM 64
#define TQ 2048
#define TK 2880   // 2048 self + 768 adapter + 64 task

// CDNA5 async global->LDS copy (ASYNCcnt-tracked), 16B per lane.
// lds: low 32 bits of flat shared pointer (== LDS byte address).
// gaddr: 64-bit global address.
__device__ __forceinline__ void async_copy_b128(unsigned lds, unsigned long long gaddr) {
    asm volatile("global_load_async_to_lds_b128 %0, %1, off"
                 :: "v"(lds), "v"(gaddr) : "memory");
}
__device__ __forceinline__ void async_copy_b128_off16(unsigned lds, unsigned long long gaddr) {
    asm volatile("global_load_async_to_lds_b128 %0, %1, off offset:16"
                 :: "v"(lds), "v"(gaddr) : "memory");
}
__device__ __forceinline__ void wait_async0() {
    asm volatile("s_wait_asynccnt 0x0" ::: "memory");
}

// ---------------------------------------------------------------------------
// f32 -> bf16 convert (plain)
// ---------------------------------------------------------------------------
__global__ void k_cvt(const float* __restrict__ in, __bf16* __restrict__ out, long n) {
    long i = (long)blockIdx.x * blockDim.x + threadIdx.x;
    if (i < n) out[i] = (__bf16)in[i];
}

// weight convert + transpose: in[k][n] f32 -> out[n][k] bf16  (1024x1024)
__global__ void k_cvt_w_t(const float* __restrict__ in, __bf16* __restrict__ out) {
    long idx = (long)blockIdx.x * blockDim.x + threadIdx.x;
    if (idx >= (long)DIM * DIM) return;
    int n = (int)(idx / DIM);
    int k = (int)(idx % DIM);
    out[idx] = (__bf16)in[(long)k * DIM + n];
}

// concat(h_a[2,512,1024], p[2,256,1024]) -> bf16 [2,768,1024]
__global__ void k_concat_adapter(const float* __restrict__ ha, const float* __restrict__ pp,
                                 __bf16* __restrict__ out) {
    long idx = (long)blockIdx.x * blockDim.x + threadIdx.x;
    const long n = 2L * 768 * DIM;
    if (idx >= n) return;
    int c = (int)(idx % DIM);
    long r = idx / DIM;
    int t = (int)(r % 768);
    int b = (int)(r / 768);
    float v = (t < 512) ? ha[((long)b * 512 + t) * DIM + c]
                        : pp[((long)b * 256 + (t - 512)) * DIM + c];
    out[idx] = (__bf16)v;
}

// ---------------------------------------------------------------------------
// bf16 WMMA GEMM:  C[M,N] = A[M,K] @ Bt[N,K]^T + bias  (+residual / ReLU)
// Bt is the PRE-TRANSPOSED weight ([N][K]).
// block = 256 threads (8 waves), tile 128x128, BK=32, double-buffered async
// global->LDS staging (1 barrier + 1 asynccnt wait per K-step).
// Wave (mw = w&3, nw = w>>2) computes rows 32*mw..+31, cols 64*nw..+63:
// 8 WMMAs per K-step per wave.
// mode: 0 = bias only, 1 = bias + residual, 2 = bias + ReLU
// ---------------------------------------------------------------------------
__global__ __launch_bounds__(256)
void k_gemm(const __bf16* __restrict__ A, const __bf16* __restrict__ Bt,
            const float* __restrict__ bias, float* __restrict__ C,
            const float* __restrict__ resid, int M, int K, int N, int mode) {
    __shared__ __align__(16) __bf16 As[2][128][40];   // [buf][m][k], 80B row stride
    __shared__ __align__(16) __bf16 Bs[2][128][40];   // [buf][n][k], 80B row stride

    const int tid  = threadIdx.x;
    const int lane = tid & 31;
    const int wave = tid >> 5;
    const int ln   = lane & 15;
    const int hl   = lane >> 4;
    const int mw   = wave & 3;
    const int nw   = wave >> 2;
    const int m0   = blockIdx.y * 128;
    const int n0   = blockIdx.x * 128;

    // staging: 128 rows x 32 k per tile; row = tid>>1, 16-elem half = (tid&1)*16
    const int sr = tid >> 1;
    const int sh = (tid & 1) * 16;
    const __bf16* gA = A  + (long)(m0 + sr) * K + sh;
    const __bf16* gB = Bt + (long)(n0 + sr) * K + sh;

    auto stage = [&](int buf, int k0) {
        unsigned la = (unsigned)(size_t)&As[buf][sr][sh];
        unsigned lb = (unsigned)(size_t)&Bs[buf][sr][sh];
        unsigned long long ga = (unsigned long long)(size_t)(gA + k0);
        unsigned long long gb = (unsigned long long)(size_t)(gB + k0);
        async_copy_b128(la, ga);
        async_copy_b128_off16(la, ga);
        async_copy_b128(lb, gb);
        async_copy_b128_off16(lb, gb);
    };

    v8f acc[2][4] = {};
    stage(0, 0);
    int buf = 0;
    for (int k0 = 0; k0 < K; k0 += 32) {
        wait_async0();       // my async fills (and issue slot) complete
        __syncthreads();     // everyone's fills visible; prev reads done
        if (k0 + 32 < K) {
            stage(buf ^ 1, k0 + 32);
            __builtin_prefetch(gA + k0 + 64, 0, 3);
            __builtin_prefetch(gB + k0 + 64, 0, 3);
        }

        // A fragments: lane m=ln holds K = 8h..8h+7 and 16+8h..16+8h+7
        BF16Frag a[2];
        #pragma unroll
        for (int am = 0; am < 2; ++am) {
            const int row = 32 * mw + 16 * am + ln;
            a[am].q[0] = *(const uint4*)&As[buf][row][8 * hl];
            a[am].q[1] = *(const uint4*)&As[buf][row][16 + 8 * hl];
        }
        // B fragments: lane n=ln holds K = 8h..8h+7 and 16+8h..16+8h+7
        #pragma unroll
        for (int j = 0; j < 4; ++j) {
            BF16Frag b;
            const int row = 64 * nw + 16 * j + ln;
            b.q[0] = *(const uint4*)&Bs[buf][row][8 * hl];
            b.q[1] = *(const uint4*)&Bs[buf][row][16 + 8 * hl];
            #pragma unroll
            for (int am = 0; am < 2; ++am) {
                acc[am][j] = __builtin_amdgcn_wmma_f32_16x16x32_bf16(
                    false, a[am].v, false, b.v, (short)0, acc[am][j], false, false);
            }
        }
        buf ^= 1;
    }

    // epilogue: C layout row = r + 8h, col = ln (+16j)
    #pragma unroll
    for (int am = 0; am < 2; ++am) {
        #pragma unroll
        for (int j = 0; j < 4; ++j) {
            #pragma unroll
            for (int r = 0; r < 8; ++r) {
                int row = m0 + 32 * mw + 16 * am + r + 8 * hl;
                int col = n0 + 64 * nw + 16 * j + ln;
                long idx = (long)row * N + col;
                float v = acc[am][j][r] + bias[col];
                if (mode == 1) v += resid[idx];
                if (mode == 2) v = fmaxf(v, 0.0f);
                C[idx] = v;
            }
        }
    }
}

// ---------------------------------------------------------------------------
// RoPE + head pack: src [b][t][1024] f32 -> dst [b][h][tstride tokens][64] bf16
// interleaved rotate-half, NeoX cos/sin tables (freq index = d & 31)
// ---------------------------------------------------------------------------
__global__ void k_rope_pack(const float* __restrict__ src, __bf16* __restrict__ dst,
                            int tstride, int koff) {
    long idx = (long)blockIdx.x * blockDim.x + threadIdx.x;  // pair index
    const long n = 2L * TQ * NHEAD * 32;
    if (idx >= n) return;
    int i2 = (int)(idx % 32);
    long r = idx / 32;
    int hh = (int)(r % NHEAD); r /= NHEAD;
    int t  = (int)(r % TQ);
    int b  = (int)(r / TQ);
    int d  = 2 * i2;

    const float* s = src + ((long)b * TQ + t) * DIM + hh * HDIM;
    float xe = s[d], xo = s[d + 1];
    float je = (float)(d & 31), jo = (float)((d + 1) & 31);
    float fe = powf(10000.0f, -je * (1.0f / 32.0f));
    float fo = powf(10000.0f, -jo * (1.0f / 32.0f));
    float tf = (float)t;
    float oe = xe * cosf(tf * fe) - xo * sinf(tf * fe);
    float oo = xo * cosf(tf * fo) + xe * sinf(tf * fo);

    __bf16* dp = dst + ((long)(b * NHEAD + hh) * tstride + koff + t) * HDIM;
    dp[d]     = (__bf16)oe;
    dp[d + 1] = (__bf16)oo;
}

// plain head pack into k_all (optional tanh(gating) scale for task keys)
__global__ void k_pack_k(const float* __restrict__ src, __bf16* __restrict__ dst,
                         int Tsrc, int koff, const float* __restrict__ gate) {
    long idx = (long)blockIdx.x * blockDim.x + threadIdx.x;
    long n = 2L * Tsrc * DIM;
    if (idx >= n) return;
    int c = (int)(idx % DIM);
    long r = idx / DIM;
    int t = (int)(r % Tsrc);
    int b = (int)(r / Tsrc);
    float v = src[idx];
    if (gate) v *= tanhf(gate[0]);
    int hh = c >> 6, d = c & 63;
    dst[((long)(b * NHEAD + hh) * TK + koff + t) * HDIM + d] = (__bf16)v;
}

// head pack V transposed: dst [b][h][d][TK] bf16
__global__ void k_pack_v(const float* __restrict__ src, __bf16* __restrict__ dst,
                         int Tsrc, int voff) {
    long idx = (long)blockIdx.x * blockDim.x + threadIdx.x;
    long n = 2L * Tsrc * DIM;
    if (idx >= n) return;
    int c = (int)(idx % DIM);
    long r = idx / DIM;
    int t = (int)(r % Tsrc);
    int b = (int)(r / Tsrc);
    int hh = c >> 6, d = c & 63;
    dst[((long)(b * NHEAD + hh) * HDIM + d) * TK + voff + t] = (__bf16)src[idx];
}

// ---------------------------------------------------------------------------
// Flash attention: grid (B*H, TQ/128), block 256 (8 waves x 16 queries).
// All 8 waves share each 32-key K/V chunk through LDS, staged with CDNA5
// async global->LDS copies (K+V is 23.6 MB total -> chunk re-reads stay
// L2-resident on the 192 MB L2).
// qp [b][h][2048][64], kall [b][h][2880][64], vtb [b][h][64][2880] (bf16)
// ao  [b][t][1024] f32
// ---------------------------------------------------------------------------
__global__ __launch_bounds__(256)
void k_attention(const __bf16* __restrict__ qp, const __bf16* __restrict__ kall,
                 const __bf16* __restrict__ vtb, float* __restrict__ ao) {
    __shared__ __align__(16) __bf16 Ks[32][72];   // [key][d], 144B row stride
    __shared__ __align__(16) __bf16 Vs[64][40];   // [d][key], 80B row stride
    __shared__ float pl[8][16][36];               // per-wave P transpose scratch

    const int tid  = threadIdx.x;
    const int lane = tid & 31;
    const int wave = tid >> 5;
    const int ln   = lane & 15;
    const int hl   = lane >> 4;
    const int bh   = blockIdx.x;
    const int b    = bh >> 4;
    const int h    = bh & 15;
    const int q0   = blockIdx.y * 128 + wave * 16;

    const __bf16* qbase = qp   + ((long)bh * TQ + q0) * HDIM;
    const __bf16* kbase = kall + (long)bh * TK * HDIM;
    const __bf16* vbase = vtb  + (long)bh * HDIM * TK;

    // staging indices: K chunk 32x64 (8 bf16 per thread), V chunk 64x32
    const int krow = tid >> 3, kseg = (tid & 7) * 8;   // Ks[krow][kseg..+7]
    const int vrow = tid >> 2, vseg = (tid & 3) * 8;   // Vs[vrow][vseg..+7]
    const unsigned ldsK = (unsigned)(size_t)&Ks[krow][kseg];
    const unsigned ldsV = (unsigned)(size_t)&Vs[vrow][vseg];

    // q A-fragments: a0 = d 0..31, a1 = d 32..63
    BF16Frag a0, a1;
    {
        const __bf16* qr = qbase + ln * HDIM;
        a0.q[0] = *(const uint4*)(qr + 8 * hl);
        a0.q[1] = *(const uint4*)(qr + 16 + 8 * hl);
        a1.q[0] = *(const uint4*)(qr + 32 + 8 * hl);
        a1.q[1] = *(const uint4*)(qr + 48 + 8 * hl);
    }

    v8f o[4] = {};
    float rmax[8], rsum[8];
    #pragma unroll
    for (int r = 0; r < 8; ++r) { rmax[r] = -__builtin_inff(); rsum[r] = 0.0f; }

    for (int kc = 0; kc < TK; kc += 32) {
        __syncthreads();   // previous chunk's LDS reads complete
        // cooperative async K/V chunk staging
        async_copy_b128(ldsK, (unsigned long long)(size_t)(kbase + (long)(kc + krow) * HDIM + kseg));
        async_copy_b128(ldsV, (unsigned long long)(size_t)(vbase + (long)vrow * TK + kc + vseg));
        if (kc + 32 < TK) {
            __builtin_prefetch(kbase + (long)(kc + 32 + krow) * HDIM + kseg, 0, 3);
            __builtin_prefetch(vbase + (long)vrow * TK + kc + 32 + vseg, 0, 3);
        }
        wait_async0();
        __syncthreads();

        // scores: s0 -> keys kc..kc+15, s1 -> keys kc+16..kc+31
        v8f s0 = {}, s1 = {};
        {
            BF16Frag b0, b1;
            b0.q[0] = *(const uint4*)&Ks[ln][8 * hl];
            b0.q[1] = *(const uint4*)&Ks[ln][16 + 8 * hl];
            b1.q[0] = *(const uint4*)&Ks[ln][32 + 8 * hl];
            b1.q[1] = *(const uint4*)&Ks[ln][48 + 8 * hl];
            s0 = __builtin_amdgcn_wmma_f32_16x16x32_bf16(false, a0.v, false, b0.v, (short)0, s0, false, false);
            s0 = __builtin_amdgcn_wmma_f32_16x16x32_bf16(false, a1.v, false, b1.v, (short)0, s0, false, false);
        }
        {
            BF16Frag b0, b1;
            b0.q[0] = *(const uint4*)&Ks[16 + ln][8 * hl];
            b0.q[1] = *(const uint4*)&Ks[16 + ln][16 + 8 * hl];
            b1.q[0] = *(const uint4*)&Ks[16 + ln][32 + 8 * hl];
            b1.q[1] = *(const uint4*)&Ks[16 + ln][48 + 8 * hl];
            s1 = __builtin_amdgcn_wmma_f32_16x16x32_bf16(false, a0.v, false, b0.v, (short)0, s1, false, false);
            s1 = __builtin_amdgcn_wmma_f32_16x16x32_bf16(false, a1.v, false, b1.v, (short)0, s1, false, false);
        }

        // online softmax (C layout: row = r + 8*hl, col = ln)
        #pragma unroll
        for (int r = 0; r < 8; ++r) {
            float x0 = s0[r] * 0.125f;   // 1/sqrt(64)
            float x1 = s1[r] * 0.125f;
            float cmax = fmaxf(x0, x1);
            cmax = fmaxf(cmax, __shfl_xor(cmax, 1));
            cmax = fmaxf(cmax, __shfl_xor(cmax, 2));
            cmax = fmaxf(cmax, __shfl_xor(cmax, 4));
            cmax = fmaxf(cmax, __shfl_xor(cmax, 8));
            float nm = fmaxf(rmax[r], cmax);
            float corr = __expf(rmax[r] - nm);
            rmax[r] = nm;
            float p0 = __expf(x0 - nm);
            float p1 = __expf(x1 - nm);
            float ps = p0 + p1;
            ps += __shfl_xor(ps, 1);
            ps += __shfl_xor(ps, 2);
            ps += __shfl_xor(ps, 4);
            ps += __shfl_xor(ps, 8);
            rsum[r] = rsum[r] * corr + ps;
            #pragma unroll
            for (int j = 0; j < 4; ++j) o[j][r] *= corr;
            pl[wave][r + 8 * hl][ln]      = p0;
            pl[wave][r + 8 * hl][16 + ln] = p1;
        }
        // P: C-layout -> A-layout (within-wave LDS RAW; compiler inserts dscnt wait)
        BF16Frag pa;
        #pragma unroll
        for (int e = 0; e < 8; ++e) {
            pa.e[e]     = (__bf16)pl[wave][ln][8 * hl + e];
            pa.e[8 + e] = (__bf16)pl[wave][ln][16 + 8 * hl + e];
        }

        // O += P @ V, V B-fragment from LDS (keys 8h..8h+7, 16+8h..16+8h+7)
        #pragma unroll
        for (int j = 0; j < 4; ++j) {
            BF16Frag vb;
            vb.q[0] = *(const uint4*)&Vs[16 * j + ln][8 * hl];
            vb.q[1] = *(const uint4*)&Vs[16 * j + ln][16 + 8 * hl];
            o[j] = __builtin_amdgcn_wmma_f32_16x16x32_bf16(false, pa.v, false, vb.v, (short)0, o[j], false, false);
        }
    }

    // finalize: divide by row sums, write [b][t][h*64+d] f32
    #pragma unroll
    for (int r = 0; r < 8; ++r) {
        float inv = 1.0f / rsum[r];
        int tok = q0 + r + 8 * hl;
        #pragma unroll
        for (int j = 0; j < 4; ++j)
            ao[((long)b * TQ + tok) * DIM + h * HDIM + 16 * j + ln] = o[j][r] * inv;
    }
}

// ---------------------------------------------------------------------------
// LayerNorm over rows of 1024, output bf16 for the final GEMM
// ---------------------------------------------------------------------------
__global__ __launch_bounds__(256)
void k_layernorm(const float* __restrict__ in, const float* __restrict__ w,
                 const float* __restrict__ bvec, __bf16* __restrict__ out) {
    __shared__ float s_sum[256];
    __shared__ float s_sq[256];
    const int row = blockIdx.x;
    const float* x = in + (long)row * DIM;
    float sum = 0.0f, sq = 0.0f;
    #pragma unroll
    for (int i = 0; i < 4; ++i) {
        float v = x[threadIdx.x + 256 * i];
        sum += v; sq += v * v;
    }
    s_sum[threadIdx.x] = sum;
    s_sq[threadIdx.x]  = sq;
    __syncthreads();
    for (int st = 128; st > 0; st >>= 1) {
        if (threadIdx.x < st) {
            s_sum[threadIdx.x] += s_sum[threadIdx.x + st];
            s_sq[threadIdx.x]  += s_sq[threadIdx.x + st];
        }
        __syncthreads();
    }
    float mu  = s_sum[0] * (1.0f / DIM);
    float var = s_sq[0] * (1.0f / DIM) - mu * mu;
    float inv = rsqrtf(var + 1e-5f);
    #pragma unroll
    for (int i = 0; i < 4; ++i) {
        int c = threadIdx.x + 256 * i;
        float v = (x[c] - mu) * inv * w[c] + bvec[c];
        out[(long)row * DIM + c] = (__bf16)v;
    }
}

// ---------------------------------------------------------------------------
// Launcher
// ---------------------------------------------------------------------------
extern "C" void kernel_launch(void* const* d_in, const int* in_sizes, int n_in,
                              void* d_out, int out_size, void* d_ws, size_t ws_size,
                              hipStream_t stream) {
    const float* x   = (const float*)d_in[0];
    const float* h_a = (const float*)d_in[1];
    const float* h_t = (const float*)d_in[2];
    const float* p   = (const float*)d_in[3];
    const float* W[9];
    const float* bW[9];
    for (int i = 0; i < 9; ++i) {
        W[i]  = (const float*)d_in[4 + 2 * i];
        bW[i] = (const float*)d_in[5 + 2 * i];
    }
    const float* gating = (const float*)d_in[22];
    const float* ln_w   = (const float*)d_in[23];
    const float* ln_b   = (const float*)d_in[24];
    float* out = (float*)d_out;

    // --- workspace carve (256B aligned) ---
    size_t off = 0;
    char* base = (char*)d_ws;
    auto take = [&](size_t bytes) -> void* {
        void* ptr = base + off;
        off += (bytes + 255) & ~(size_t)255;
        return ptr;
    };
    __bf16* Wb[9];   // transposed [N][K] bf16 weights
    for (int i = 0; i < 9; ++i) Wb[i] = (__bf16*)take(1024L * 1024 * 2);
    __bf16* xb  = (__bf16*)take(4096L * 1024 * 2);
    __bf16* hab = (__bf16*)take(1536L * 1024 * 2);
    __bf16* htb = (__bf16*)take(128L * 1024 * 2);
    float* qf  = (float*)take(4096L * 1024 * 4);
    float* ksf = (float*)take(4096L * 1024 * 4);
    float* vsf = (float*)take(4096L * 1024 * 4);
    float* kaf = (float*)take(1536L * 1024 * 4);
    float* vaf = (float*)take(1536L * 1024 * 4);
    float* ktf = (float*)take(128L * 1024 * 4);
    float* vtf = (float*)take(128L * 1024 * 4);
    __bf16* qp   = (__bf16*)take(2L * 16 * 2048 * 64 * 2);
    __bf16* kall = (__bf16*)take(2L * 16 * 2880 * 64 * 2);
    __bf16* vtb  = (__bf16*)take(2L * 16 * 2880 * 64 * 2);
    // safe aliases (buffers dead by time of reuse)
    float*  ao   = qf;               // attention output f32
    __bf16* aob  = xb;               // attention output bf16
    float*  res1 = ksf;              // Wo output + residual
    __bf16* lnb  = (__bf16*)vsf;     // layernorm output bf16

    auto cv = [&](const float* src, __bf16* dst, long n) {
        k_cvt<<<dim3((unsigned)((n + 255) / 256)), dim3(256), 0, stream>>>(src, dst, n);
    };
    // 1) convert weights (transposed) + activations to bf16
    for (int i = 0; i < 9; ++i)
        k_cvt_w_t<<<dim3((unsigned)((1024L * 1024 + 255) / 256)), dim3(256), 0, stream>>>(W[i], Wb[i]);
    cv(x, xb, 4096L * 1024);
    k_concat_adapter<<<dim3((unsigned)((2L * 768 * 1024 + 255) / 256)), dim3(256), 0, stream>>>(h_a, p, hab);
    cv(h_t, htb, 128L * 1024);

    auto gemm = [&](const __bf16* A, const __bf16* Bw, const float* bias, float* C,
                    const float* resid, int M, int mode) {
        dim3 g(1024 / 128, (unsigned)(M / 128));
        k_gemm<<<g, dim3(256), 0, stream>>>(A, Bw, bias, C, resid, M, 1024, 1024, mode);
    };
    // 2) projections
    gemm(xb,  Wb[0], bW[0], qf,  nullptr, 4096, 0);  // Wq
    gemm(xb,  Wb[1], bW[1], ksf, nullptr, 4096, 0);  // Wks
    gemm(xb,  Wb[2], bW[2], vsf, nullptr, 4096, 0);  // Wvs
    gemm(hab, Wb[3], bW[3], kaf, nullptr, 1536, 0);  // Wka
    gemm(hab, Wb[4], bW[4], vaf, nullptr, 1536, 0);  // Wva
    gemm(htb, Wb[5], bW[5], ktf, nullptr, 128,  0);  // Wkt
    gemm(htb, Wb[6], bW[6], vtf, nullptr, 128,  0);  // Wvt

    // 3) RoPE + pack q/k/v
    long nq = 2L * 2048 * 16 * 32;
    k_rope_pack<<<dim3((unsigned)((nq + 255) / 256)), dim3(256), 0, stream>>>(qf,  qp,   2048, 0);
    k_rope_pack<<<dim3((unsigned)((nq + 255) / 256)), dim3(256), 0, stream>>>(ksf, kall, 2880, 0);
    k_pack_k<<<dim3((unsigned)((2L * 768 * 1024 + 255) / 256)), dim3(256), 0, stream>>>(kaf, kall, 768, 2048, nullptr);
    k_pack_k<<<dim3((unsigned)((2L * 64 * 1024 + 255) / 256)),  dim3(256), 0, stream>>>(ktf, kall, 64, 2816, gating);
    k_pack_v<<<dim3((unsigned)((2L * 2048 * 1024 + 255) / 256)), dim3(256), 0, stream>>>(vsf, vtb, 2048, 0);
    k_pack_v<<<dim3((unsigned)((2L * 768 * 1024 + 255) / 256)),  dim3(256), 0, stream>>>(vaf, vtb, 768, 2048);
    k_pack_v<<<dim3((unsigned)((2L * 64 * 1024 + 255) / 256)),   dim3(256), 0, stream>>>(vtf, vtb, 64, 2816);

    // 4) attention
    k_attention<<<dim3(32, 16), dim3(256), 0, stream>>>(qp, kall, vtb, ao);

    // 5) out proj (+residual), layernorm, FFN (+ReLU)
    cv(ao, aob, 4096L * 1024);
    gemm(aob, Wb[7], bW[7], res1, x, 4096, 1);       // Wo + residual x
    k_layernorm<<<dim3(4096), dim3(256), 0, stream>>>(res1, ln_w, ln_b, lnb);
    gemm(lnb, Wb[8], bW[8], out, nullptr, 4096, 2);  // Wf + ReLU -> d_out
}
